// AxialAttention_61323543052759
// MI455X (gfx1250) — compile-verified, matrix operand order checked by reading, not run on hardware
//
#include <hip/hip_runtime.h>

typedef __attribute__((ext_vector_type(16))) __bf16 bf16x16;
typedef __attribute__((ext_vector_type(8)))  float  floatx8;
typedef unsigned short u16;

#define C_   256
#define H_   64
#define W_   64
#define L_   64      // sequence length (width)
#define HEADS_ 8
#define DH_  32      // head dim
#define QS_  264     // LDS row stride (halves): 528B rows keep 16B-aligned b128 frags
#define PS_  72      // LDS row stride (halves) for 64-wide P buffer
#define OS_  66      // LDS row stride (floats) for output staging

#define WQKV_T_ELEMS (768 * 256)
#define WPROJ_T_ELEMS (256 * 256)

__device__ __forceinline__ u16 f2bfu(float f) {
  unsigned u = __builtin_bit_cast(unsigned, f);
  u += 0x7FFFu + ((u >> 16) & 1u);          // round-to-nearest-even
  return (u16)(u >> 16);
}
__device__ __forceinline__ __bf16 bfb(u16 u) { return __builtin_bit_cast(__bf16, u); }

__device__ __forceinline__ floatx8 wmma_bf16(bf16x16 a, bf16x16 b, floatx8 c) {
  return __builtin_amdgcn_wmma_f32_16x16x32_bf16(false, a, false, b, (short)0, c,
                                                 false, false);
}

// 16x32 bf16 fragment where the 32 K-values are contiguous per row.
// Works for: A from row-major [M][K]; B = X^T from row-major [N][K];
// LDS (ds_load_b128 x2) or global (global_load_b128 x2) sources alike.
__device__ __forceinline__ bf16x16 frag_rowK(const u16* p, int stride, int row0,
                                             int k0, int lane16, int hi) {
  const u16* r = p + (row0 + lane16) * stride + k0 + hi * 8;
  bf16x16 a;
#pragma unroll
  for (int e = 0; e < 8; ++e) { a[e] = bfb(r[e]); a[8 + e] = bfb(r[16 + e]); }
  return a;
}

// B fragment (32x16 bf16) from row-major [K][N] LDS source (column-strided).
__device__ __forceinline__ bf16x16 frag_colK_lds(const u16* p, int stride, int k0,
                                                 int col0, int lane16, int hi) {
  const u16* r = p + (k0 + hi * 8) * stride + col0 + lane16;
  bf16x16 b;
#pragma unroll
  for (int e = 0; e < 8; ++e) {
    b[e]     = bfb(r[e * stride]);
    b[8 + e] = bfb(r[(16 + e) * stride]);
  }
  return b;
}

// Fallback: B fragment from row-major [K][N] global fp32, convert on the fly.
__device__ __forceinline__ bf16x16 frag_colK_gf32(const float* p, int stride, int k0,
                                                  int col0, int lane16, int hi) {
  const float* r = p + (k0 + hi * 8) * stride + col0 + lane16;
  bf16x16 b;
#pragma unroll
  for (int e = 0; e < 8; ++e) {
    b[e]     = bfb(f2bfu(r[e * stride]));
    b[8 + e] = bfb(f2bfu(r[(16 + e) * stride]));
  }
  return b;
}

// ---- Pre-pass: transpose + convert weights to bf16 [N][K] in workspace ----
extern "C" __global__ __launch_bounds__(256)
void weights_to_bf16T(const float* __restrict__ wqkv, const float* __restrict__ wproj,
                      u16* __restrict__ wqkvT, u16* __restrict__ wprojT) {
  const int blk = blockIdx.x, t = threadIdx.x;
  if (blk < 768) {
    wqkvT[blk * 256 + t] = f2bfu(wqkv[t * 768 + blk]);   // [768][256] <- [256][768]^T
  } else {
    const int n = blk - 768;
    wprojT[n * 256 + t] = f2bfu(wproj[t * 256 + n]);     // [256][256] <- ^T
  }
}

extern "C" __global__ __launch_bounds__(256, 1)
void axial_attn_fused(const float* __restrict__ x,
                      const float* __restrict__ rbias,
                      const float* __restrict__ wqkv,
                      const float* __restrict__ bqkv,
                      const float* __restrict__ wproj,
                      const float* __restrict__ bproj,
                      const u16* __restrict__ wqkvT,
                      const u16* __restrict__ wprojT,
                      const int useTw,
                      float* __restrict__ out) {
  extern __shared__ char smem[];
  u16* qBuf = (u16*)smem;                 // [64][QS_] bf16  (later reused for O)
  u16* kBuf = qBuf + L_ * QS_;            // [64][QS_] bf16
  u16* vBuf = kBuf + L_ * QS_;            // [64][QS_] bf16
  char* U   = (char*)(vBuf + L_ * QS_);   // 73728-byte union region
  u16*   seqB = (u16*)U;                  // [64][QS_]          (phase A/B)
  u16*   pBuf = (u16*)U;                  // [8][64][PS_]       (phase C/D)
  float* oBuf = (float*)U;                // [256][OS_]         (phase E/F)

  const int t = threadIdx.x;
  const int wv = t >> 5, lane = t & 31, lane16 = lane & 15, hi = lane >> 4;
  const int n  = blockIdx.x;              // sequence id in [0, B*T*H)
  const int h  = n & (H_ - 1);
  const int bt = n >> 6;
  const long base = (long)bt * C_ * (H_ * W_) + (long)h * W_;

  const floatx8 zero = {0.f, 0.f, 0.f, 0.f, 0.f, 0.f, 0.f, 0.f};

  // ---- Phase A: x slice [C][W] -> seq[W][C] bf16 (transpose into LDS) ----
  {
    const float* xrow = x + base + (long)t * (H_ * W_);   // channel c = t
#pragma unroll
    for (int w4 = 0; w4 < 16; ++w4) {
      float4 f = ((const float4*)xrow)[w4];
      int w = w4 * 4;
      seqB[(w + 0) * QS_ + t] = f2bfu(f.x);
      seqB[(w + 1) * QS_ + t] = f2bfu(f.y);
      seqB[(w + 2) * QS_ + t] = f2bfu(f.z);
      seqB[(w + 3) * QS_ + t] = f2bfu(f.w);
    }
  }
  __syncthreads();

  // ---- Phase B: qkv = seq @ Wqkv + b  (wave wv owns 96 of 768 columns) ----
  for (int nt = 0; nt < 6; ++nt) {
    const int ncol0 = wv * 96 + nt * 16;
    floatx8 acc[4] = {zero, zero, zero, zero};
    if (useTw) {
      for (int k0 = 0; k0 < C_; k0 += 32) {
        // B^T is row-K contiguous: two global_load_b128 per lane, no conversion
        bf16x16 bf = frag_rowK(wqkvT, C_, ncol0, k0, lane16, hi);
#pragma unroll
        for (int mi = 0; mi < 4; ++mi) {
          bf16x16 af = frag_rowK(seqB, QS_, mi * 16, k0, lane16, hi);
          acc[mi] = wmma_bf16(af, bf, acc[mi]);
        }
      }
    } else {
      for (int k0 = 0; k0 < C_; k0 += 32) {
        bf16x16 bf = frag_colK_gf32(wqkv, 3 * C_, k0, ncol0, lane16, hi);
#pragma unroll
        for (int mi = 0; mi < 4; ++mi) {
          bf16x16 af = frag_rowK(seqB, QS_, mi * 16, k0, lane16, hi);
          acc[mi] = wmma_bf16(af, bf, acc[mi]);
        }
      }
    }
    const float bias = bqkv[ncol0 + lane16];
    u16* dst = (ncol0 < 256) ? qBuf : (ncol0 < 512 ? kBuf : vBuf);
    const int lcol = (ncol0 & 255) + lane16;
#pragma unroll
    for (int mi = 0; mi < 4; ++mi)
#pragma unroll
      for (int v = 0; v < 8; ++v) {
        int row = mi * 16 + v + hi * 8;
        dst[row * QS_ + lcol] = f2bfu(acc[mi][v] + bias);
      }
  }
  __syncthreads();

  // ---- Phase C: per-head S = scale*q k^T + bias, row softmax -> P ----
  const int hd = wv, hc0 = hd * DH_;
  u16* P = pBuf + hd * (L_ * PS_);
  const float* bh = rbias + hd * (L_ * L_);
  const float scale = 0.1767766952966369f;   // 1/sqrt(32)
  bf16x16 bk[4];
#pragma unroll
  for (int ni = 0; ni < 4; ++ni)
    bk[ni] = frag_rowK(kBuf, QS_, ni * 16, hc0, lane16, hi);  // B = K^T

  for (int mi = 0; mi < 4; ++mi) {
    bf16x16 aq = frag_rowK(qBuf, QS_, mi * 16, hc0, lane16, hi);
    floatx8 s[4];
#pragma unroll
    for (int ni = 0; ni < 4; ++ni) s[ni] = wmma_bf16(aq, bk[ni], zero);
#pragma unroll
    for (int ni = 0; ni < 4; ++ni)
#pragma unroll
      for (int v = 0; v < 8; ++v) {
        int row = mi * 16 + v + hi * 8, col = ni * 16 + lane16;
        s[ni][v] = s[ni][v] * scale + bh[row * L_ + col];
      }
    // softmax over 64 columns of each row (owned by one 16-lane half, wave32)
#pragma unroll
    for (int v = 0; v < 8; ++v) {
      float mx = fmaxf(fmaxf(s[0][v], s[1][v]), fmaxf(s[2][v], s[3][v]));
#pragma unroll
      for (int off = 1; off <= 8; off <<= 1) mx = fmaxf(mx, __shfl_xor(mx, off, 32));
      float sm = 0.f;
#pragma unroll
      for (int ni = 0; ni < 4; ++ni) {
        float e = __expf(s[ni][v] - mx);
        s[ni][v] = e;
        sm += e;
      }
#pragma unroll
      for (int off = 1; off <= 8; off <<= 1) sm += __shfl_xor(sm, off, 32);
      float inv = 1.0f / sm;
#pragma unroll
      for (int ni = 0; ni < 4; ++ni) s[ni][v] *= inv;
    }
#pragma unroll
    for (int ni = 0; ni < 4; ++ni)
#pragma unroll
      for (int v = 0; v < 8; ++v)
        P[(mi * 16 + v + hi * 8) * PS_ + ni * 16 + lane16] = f2bfu(s[ni][v]);
  }
  __syncthreads();

  // ---- Phase D: O = P @ V per head; O (bf16) overwrites qBuf head columns ----
  bf16x16 bv[2][2];
#pragma unroll
  for (int kk = 0; kk < 2; ++kk)
#pragma unroll
    for (int ni = 0; ni < 2; ++ni)
      bv[kk][ni] = frag_colK_lds(vBuf, QS_, kk * 32, hc0 + ni * 16, lane16, hi);
  for (int mi = 0; mi < 4; ++mi) {
    bf16x16 a0 = frag_rowK(P, PS_, mi * 16, 0,  lane16, hi);
    bf16x16 a1 = frag_rowK(P, PS_, mi * 16, 32, lane16, hi);
#pragma unroll
    for (int ni = 0; ni < 2; ++ni) {
      floatx8 o = wmma_bf16(a1, bv[1][ni], wmma_bf16(a0, bv[0][ni], zero));
#pragma unroll
      for (int v = 0; v < 8; ++v)
        qBuf[(mi * 16 + v + hi * 8) * QS_ + hc0 + ni * 16 + lane16] = f2bfu(o[v]);
    }
  }
  __syncthreads();

  // ---- Phase E: out = O @ Wproj + b, staged transposed in oBuf[C][OS_] ----
  for (int nt = 0; nt < 2; ++nt) {
    const int ncol0 = wv * 32 + nt * 16;
    floatx8 acc[4] = {zero, zero, zero, zero};
    if (useTw) {
      for (int k0 = 0; k0 < C_; k0 += 32) {
        bf16x16 bf = frag_rowK(wprojT, C_, ncol0, k0, lane16, hi);
#pragma unroll
        for (int mi = 0; mi < 4; ++mi) {
          bf16x16 af = frag_rowK(qBuf, QS_, mi * 16, k0, lane16, hi);
          acc[mi] = wmma_bf16(af, bf, acc[mi]);
        }
      }
    } else {
      for (int k0 = 0; k0 < C_; k0 += 32) {
        bf16x16 bf = frag_colK_gf32(wproj, C_, k0, ncol0, lane16, hi);
#pragma unroll
        for (int mi = 0; mi < 4; ++mi) {
          bf16x16 af = frag_rowK(qBuf, QS_, mi * 16, k0, lane16, hi);
          acc[mi] = wmma_bf16(af, bf, acc[mi]);
        }
      }
    }
    const float bias = bproj[ncol0 + lane16];
#pragma unroll
    for (int mi = 0; mi < 4; ++mi)
#pragma unroll
      for (int v = 0; v < 8; ++v) {
        int row = mi * 16 + v + hi * 8, col = ncol0 + lane16;
        oBuf[col * OS_ + row] = acc[mi][v] + bias;
      }
  }
  __syncthreads();

  // ---- Phase F: coalesced f32 store back to [B,T,C,H,W] (row per channel) ----
  {
    float* orow = out + base + (long)t * (H_ * W_);    // channel c = t
    const float* src = oBuf + t * OS_;
#pragma unroll
    for (int w4 = 0; w4 < 16; ++w4) {
      float4 f;
      f.x = src[w4 * 4 + 0];
      f.y = src[w4 * 4 + 1];
      f.z = src[w4 * 4 + 2];
      f.w = src[w4 * 4 + 3];
      ((float4*)orow)[w4] = f;
    }
  }
}

extern "C" void kernel_launch(void* const* d_in, const int* in_sizes, int n_in,
                              void* d_out, int out_size, void* d_ws, size_t ws_size,
                              hipStream_t stream) {
  const float* x     = (const float*)d_in[0];
  const float* rbias = (const float*)d_in[1];
  const float* wqkv  = (const float*)d_in[2];
  const float* bqkv  = (const float*)d_in[3];
  const float* wproj = (const float*)d_in[4];
  const float* bproj = (const float*)d_in[5];
  float* out = (float*)d_out;

  // Workspace plan: bf16-transposed weights (deterministic: ws_size is fixed).
  const size_t needWs = (size_t)(WQKV_T_ELEMS + WPROJ_T_ELEMS) * sizeof(u16);
  const int useTw = (d_ws != nullptr && ws_size >= needWs) ? 1 : 0;
  u16* wqkvT  = (u16*)d_ws;
  u16* wprojT = wqkvT + WQKV_T_ELEMS;

  if (useTw) {
    weights_to_bf16T<<<dim3(768 + 256), dim3(256), 0, stream>>>(wqkv, wproj,
                                                                wqkvT, wprojT);
  }

  // q/k/v bf16 buffers + union(seq / P / out-stage) = 101376 + 73728 bytes
  const size_t shmem = (size_t)(3 * L_ * QS_) * sizeof(u16) + 73728;
  hipFuncSetAttribute((const void*)axial_attn_fused,
                      hipFuncAttributeMaxDynamicSharedMemorySize, (int)shmem);
  axial_attn_fused<<<dim3(4096), dim3(256), shmem, stream>>>(
      x, rbias, wqkv, bqkv, wproj, bproj, wqkvT, wprojT, useTw, out);
}
